// MaskedTripletLoss_59322088292359
// MI455X (gfx1250) — matrix operand order repaired and checked by reading, not compiled, not run on hardware
//
#include <hip/hip_runtime.h>
#include <hip/hip_bf16.h>

typedef __bf16 v16bf __attribute__((ext_vector_type(16)));
typedef float  v8f   __attribute__((ext_vector_type(8)));
typedef int    v4i   __attribute__((ext_vector_type(4)));

typedef __attribute__((address_space(1))) v4i gv4i;   // global
typedef __attribute__((address_space(3))) v4i lv4i;   // LDS

#define T_FRAMES   2048
#define D_DIM      512
#define NCODES     4096
#define B_BATCH    8
#define MARGIN_F   0.2f
#define ENC_STRIDE 320
#define NTILES     (NCODES / 16)
#define TILE_BYTES (16 * D_DIM * 2)          // 16 KB of bf16 per 16-code tile
#define CB_OFF     32768                     // byte offset of packed codebook in ws
#define WS_NEED    ((size_t)CB_OFF + (size_t)NCODES * D_DIM * 2)

#if __has_builtin(__builtin_amdgcn_global_load_async_to_lds_b128)
#define HAVE_ASYNC_LDS 1
#else
#define HAVE_ASYNC_LDS 0
#endif

static __device__ __forceinline__ __bf16 f2bf(float f) {
    unsigned u = __builtin_bit_cast(unsigned, f);
    u += 0x7FFFu + ((u >> 16) & 1u);            // round-to-nearest-even
    unsigned short s = (unsigned short)(u >> 16);
    return __builtin_bit_cast(__bf16, s);
}

static __device__ __forceinline__ unsigned pack_bf2(float lo, float hi) {
    unsigned short a = __builtin_bit_cast(unsigned short, f2bf(lo));
    unsigned short b = __builtin_bit_cast(unsigned short, f2bf(hi));
    return (unsigned)a | ((unsigned)b << 16);
}

struct BPair { uint4 a, b; };   // 32 bytes, bit-cast target for v16bf frags

// ---------------------------------------------------------------------------
// Kernel 1: ||c||^2 per code (wave per code, coalesced), zero the accumulator.
// ws layout: [0..4095]=cnorm f32, [4096]=numerator, packed bf16 cb at +32768 B.
// ---------------------------------------------------------------------------
__global__ void cnorm_kernel(const float* __restrict__ codebook,
                             float* __restrict__ ws) {
    int wave = (int)((blockIdx.x * blockDim.x + threadIdx.x) >> 5);
    int lane = (int)(threadIdx.x & 31);
    if (wave < NCODES) {
        const float* row = codebook + (size_t)wave * D_DIM;
        float s = 0.f;
        #pragma unroll
        for (int k = 0; k < D_DIM / 32; ++k) {
            float v = row[lane + k * 32];
            s += v * v;
        }
        #pragma unroll
        for (int m = 16; m >= 1; m >>= 1) s += __shfl_xor(s, m, 32);
        if (lane == 0) ws[wave] = s;
    }
    if (blockIdx.x == 0 && threadIdx.x == 0) ws[NCODES] = 0.f;
}

// ---------------------------------------------------------------------------
// Kernel 1b: pack codebook f32 -> bf16 in per-lane B-fragment order.
// For tile ct, fragment chunk kc, lane L: 16 halves = code (ct*16 + L&15),
// K = kc*32 + (L>=16)*16 + e  (contiguous k order, same as the A fragments).
// ---------------------------------------------------------------------------
__global__ void pack_codebook_kernel(const float* __restrict__ codebook,
                                     unsigned* __restrict__ cb_bf) {
    int gid  = (int)(blockIdx.x * blockDim.x + threadIdx.x);
    int ct   = gid >> 9;
    int rem  = gid & 511;
    int kc   = rem >> 5;
    int lane = rem & 31;
    int col  = (ct << 4) + (lane & 15);
    int kb   = kc * 32 + ((lane >> 4) << 4);
    const float4* src = (const float4*)(codebook + (size_t)col * D_DIM + kb);
    unsigned* dst = cb_bf + (size_t)ct * 4096 + (size_t)(kc * 32 + lane) * 8;
    #pragma unroll
    for (int j = 0; j < 4; ++j) {
        float4 v = src[j];
        dst[2 * j + 0] = pack_bf2(v.x, v.y);
        dst[2 * j + 1] = pack_bf2(v.z, v.w);
    }
}

// ---------------------------------------------------------------------------
// Double-buffered tile staging: 16 KB/tile, 64 B per thread.
// Async path: global_load_async_to_lds_b128 (ASYNCcnt); IOFFSET is applied to
// both the global and LDS address, so one base pair + imm offsets suffices.
// ---------------------------------------------------------------------------
struct Stage { uint4 d[4]; };

static __device__ __forceinline__ void stage_issue(const char* cb, int ct,
                                                   __bf16* ldsbuf, int tid,
                                                   Stage& st) {
#if HAVE_ASYNC_LDS
    (void)st;
    gv4i* src = (gv4i*)(cb + (size_t)ct * TILE_BYTES + tid * 64);
    lv4i* dst = (lv4i*)((char*)ldsbuf + tid * 64);
    __builtin_amdgcn_global_load_async_to_lds_b128(src, dst, 0,  0);
    __builtin_amdgcn_global_load_async_to_lds_b128(src, dst, 16, 0);
    __builtin_amdgcn_global_load_async_to_lds_b128(src, dst, 32, 0);
    __builtin_amdgcn_global_load_async_to_lds_b128(src, dst, 48, 0);
#else
    (void)ldsbuf;
    const uint4* src = (const uint4*)(cb + (size_t)ct * TILE_BYTES + tid * 64);
    #pragma unroll
    for (int i = 0; i < 4; ++i) st.d[i] = src[i];
#endif
}

static __device__ __forceinline__ void stage_finish(__bf16* ldsbuf, int tid,
                                                    Stage& st) {
#if HAVE_ASYNC_LDS
    (void)ldsbuf; (void)tid; (void)st;
#if __has_builtin(__builtin_amdgcn_s_wait_asynccnt)
    __builtin_amdgcn_s_wait_asynccnt(0);
#else
    asm volatile("s_wait_asynccnt 0x0" ::: "memory");
#endif
#else
    uint4* dst = (uint4*)((char*)ldsbuf + tid * 64);
    #pragma unroll
    for (int i = 0; i < 4; ++i) dst[i] = st.d[i];
#endif
}

// ---------------------------------------------------------------------------
// Kernel 2 (fast path): bf16 WMMA GEMM from pre-packed codebook with
// double-buffered (async) LDS staging + fused min/pos triplet reduction.
// ---------------------------------------------------------------------------
__launch_bounds__(256, 1)
__global__ void triplet_fast(const float* __restrict__ student,   // (B, D, T)
                             const int*   __restrict__ teacher,   // (B, T)
                             const int*   __restrict__ lengths,   // (B,)
                             const float* __restrict__ ws_cn,     // cnorm[4096]
                             const char*  __restrict__ cb_bf,     // packed bf16
                             float* __restrict__ accum) {
    __shared__ alignas(16) __bf16 lds_b2[2][16 * D_DIM];

    const int tid  = (int)threadIdx.x;
    const int wave = tid >> 5;
    const int lane = tid & 31;
    const int hi   = (lane >> 4) & 1;
    const int mrow = lane & 15;

    const int rowBase = (int)blockIdx.x * 128 + wave * 16;
    const int b     = rowBase / T_FRAMES;
    const int tbase = rowBase % T_FRAMES;

    // prologue: start staging tile 0 before the (long) A-panel load
    Stage st;
    stage_issue(cb_bf, 0, &lds_b2[0][0], tid, st);

    // ---- resident A panel (16 rows x 512 K) + z-norms ----
    v16bf a_frag[16];
    float zpart = 0.f;
    const float* zbase = student + (size_t)b * D_DIM * T_FRAMES + (tbase + mrow);
    #pragma unroll
    for (int kc = 0; kc < 16; ++kc) {
        v16bf a;
        #pragma unroll
        for (int e = 0; e < 16; ++e) {
            int k = kc * 32 + e + (hi << 4);
            float x = zbase[(size_t)k * T_FRAMES];
            zpart += x * x;
            a[e] = f2bf(x);
        }
        a_frag[kc] = a;
    }
    float znorm_m = zpart + __shfl_xor(zpart, 16, 32);
    float znr[8];
    #pragma unroll
    for (int r = 0; r < 8; ++r) znr[r] = __shfl(znorm_m, hi * 8 + r, 32);

    int tc[8];
    #pragma unroll
    for (int r = 0; r < 8; ++r) tc[r] = teacher[rowBase + hi * 8 + r];

    float qneg[8], qpos[8];
    #pragma unroll
    for (int r = 0; r < 8; ++r) { qneg[r] = 3.0e38f; qpos[r] = 3.0e38f; }

    stage_finish(&lds_b2[0][0], tid, st);
    __syncthreads();

    for (int ct = 0; ct < NTILES; ++ct) {
        const int buf = ct & 1;
        if (ct + 1 < NTILES) stage_issue(cb_bf, ct + 1, &lds_b2[buf ^ 1][0], tid, st);

        // 16 WMMAs over K=512 ; per-lane frag = 32 contiguous bytes
        v8f acc = {};
        const char* bbase = (const char*)&lds_b2[buf][0] + lane * 32;
        #pragma unroll
        for (int kc = 0; kc < 16; ++kc) {
            const uint4* p = (const uint4*)(bbase + kc * 1024);
            BPair raw; raw.a = p[0]; raw.b = p[1];
            v16bf bf = __builtin_bit_cast(v16bf, raw);
            acc = __builtin_amdgcn_wmma_f32_16x16x32_bf16(
                      false, a_frag[kc], false, bf, (short)0, acc, false, false);
        }

        const int col = ct * 16 + mrow;
        const float c2 = ws_cn[col];
        #pragma unroll
        for (int r = 0; r < 8; ++r) {
            float q = c2 - 2.0f * acc[r];
            bool isPos = (col == tc[r]);
            qpos[r] = (isPos  && q < qpos[r]) ? q : qpos[r];
            qneg[r] = (!isPos && q < qneg[r]) ? q : qneg[r];
        }

        if (ct + 1 < NTILES) stage_finish(&lds_b2[buf ^ 1][0], tid, st);
        __syncthreads();
    }

    #pragma unroll
    for (int r = 0; r < 8; ++r) {
        #pragma unroll
        for (int m = 1; m <= 8; m <<= 1) {
            qpos[r] = fminf(qpos[r], __shfl_xor(qpos[r], m, 32));
            qneg[r] = fminf(qneg[r], __shfl_xor(qneg[r], m, 32));
        }
    }

    const int frame_len = lengths[b] / ENC_STRIDE;
    float local = 0.f;
    #pragma unroll
    for (int r = 0; r < 8; ++r) {
        int t = tbase + hi * 8 + r;
        if (t < frame_len) {
            float pd = sqrtf(fmaxf(znr[r] + qpos[r], 1e-12f));
            float nd = sqrtf(fmaxf(znr[r] + qneg[r], 1e-12f));
            float trip = pd - nd + MARGIN_F;
            local += (trip > 0.f) ? trip : 0.f;
        }
    }
    float half_sum = (mrow == 0) ? local : 0.f;
    half_sum += __shfl_xor(half_sum, 16, 32);
    if (lane == 0) atomicAdd(accum, half_sum);
}

// ---------------------------------------------------------------------------
// Kernel 2 (fallback, small ws): self-converting version.
// ---------------------------------------------------------------------------
__launch_bounds__(256, 1)
__global__ void triplet_fallback(const float* __restrict__ student,
                                 const int*   __restrict__ teacher,
                                 const float* __restrict__ codebook,
                                 const int*   __restrict__ lengths,
                                 float* __restrict__ ws) {
    constexpr int KP = D_DIM + 8;
    __shared__ __bf16 lds_b[16 * KP];

    const int tid  = (int)threadIdx.x;
    const int wave = tid >> 5;
    const int lane = tid & 31;
    const int hi   = (lane >> 4) & 1;
    const int mrow = lane & 15;

    const int rowBase = (int)blockIdx.x * 128 + wave * 16;
    const int b     = rowBase / T_FRAMES;
    const int tbase = rowBase % T_FRAMES;

    v16bf a_frag[16];
    float zpart = 0.f;
    const float* zbase = student + (size_t)b * D_DIM * T_FRAMES + (tbase + mrow);
    #pragma unroll
    for (int kc = 0; kc < 16; ++kc) {
        v16bf a;
        #pragma unroll
        for (int e = 0; e < 16; ++e) {
            int k = kc * 32 + e + (hi << 4);
            float x = zbase[(size_t)k * T_FRAMES];
            zpart += x * x;
            a[e] = f2bf(x);
        }
        a_frag[kc] = a;
    }
    float znorm_m = zpart + __shfl_xor(zpart, 16, 32);
    float znr[8];
    #pragma unroll
    for (int r = 0; r < 8; ++r) znr[r] = __shfl(znorm_m, hi * 8 + r, 32);

    int tc[8];
    #pragma unroll
    for (int r = 0; r < 8; ++r) tc[r] = teacher[rowBase + hi * 8 + r];

    float qneg[8], qpos[8];
    #pragma unroll
    for (int r = 0; r < 8; ++r) { qneg[r] = 3.0e38f; qpos[r] = 3.0e38f; }

    const int cload = tid >> 4;
    const int kload = (tid & 15) * 32;
    for (int ct = 0; ct < NTILES; ++ct) {
        const int col0 = ct * 16;
        __syncthreads();
        {
            const float4* src = (const float4*)(codebook +
                                 (size_t)(col0 + cload) * D_DIM + kload);
            unsigned* dst = (unsigned*)(lds_b + cload * KP + kload);
            #pragma unroll
            for (int i = 0; i < 8; ++i) {
                float4 v = src[i];
                dst[i * 2 + 0] = pack_bf2(v.x, v.y);
                dst[i * 2 + 1] = pack_bf2(v.z, v.w);
            }
            if (ct + 1 < NTILES)
                __builtin_prefetch((const void*)(src + 8 * (D_DIM * 16 / 32)), 0, 0);
        }
        __syncthreads();

        v8f acc = {};
        const __bf16* bptr = lds_b + mrow * KP + (hi << 4);
        #pragma unroll
        for (int kc = 0; kc < 16; ++kc) {
            const uint4* p = (const uint4*)(bptr + kc * 32);
            BPair raw; raw.a = p[0]; raw.b = p[1];
            v16bf bf = __builtin_bit_cast(v16bf, raw);
            acc = __builtin_amdgcn_wmma_f32_16x16x32_bf16(
                      false, a_frag[kc], false, bf, (short)0, acc, false, false);
        }

        const int col = col0 + mrow;
        const float c2 = ws[col];
        #pragma unroll
        for (int r = 0; r < 8; ++r) {
            float q = c2 - 2.0f * acc[r];
            bool isPos = (col == tc[r]);
            qpos[r] = (isPos  && q < qpos[r]) ? q : qpos[r];
            qneg[r] = (!isPos && q < qneg[r]) ? q : qneg[r];
        }
    }

    #pragma unroll
    for (int r = 0; r < 8; ++r) {
        #pragma unroll
        for (int m = 1; m <= 8; m <<= 1) {
            qpos[r] = fminf(qpos[r], __shfl_xor(qpos[r], m, 32));
            qneg[r] = fminf(qneg[r], __shfl_xor(qneg[r], m, 32));
        }
    }

    const int frame_len = lengths[b] / ENC_STRIDE;
    float local = 0.f;
    #pragma unroll
    for (int r = 0; r < 8; ++r) {
        int t = tbase + hi * 8 + r;
        if (t < frame_len) {
            float pd = sqrtf(fmaxf(znr[r] + qpos[r], 1e-12f));
            float nd = sqrtf(fmaxf(znr[r] + qneg[r], 1e-12f));
            float trip = pd - nd + MARGIN_F;
            local += (trip > 0.f) ? trip : 0.f;
        }
    }
    float half_sum = (mrow == 0) ? local : 0.f;
    half_sum += __shfl_xor(half_sum, 16, 32);
    if (lane == 0) atomicAdd(&ws[NCODES], half_sum);
}

// ---------------------------------------------------------------------------
// Kernel 3: loss = numerator / (sum(mask) + 1e-8)
// ---------------------------------------------------------------------------
__global__ void finalize_kernel(const int* __restrict__ lengths,
                                const float* __restrict__ ws,
                                float* __restrict__ out) {
    if (blockIdx.x == 0 && threadIdx.x == 0) {
        float den = 0.f;
        for (int i = 0; i < B_BATCH; ++i) {
            int fl = lengths[i] / ENC_STRIDE;
            if (fl > T_FRAMES) fl = T_FRAMES;
            den += (float)fl;
        }
        out[0] = ws[NCODES] / (den + 1e-8f);
    }
}

extern "C" void kernel_launch(void* const* d_in, const int* in_sizes, int n_in,
                              void* d_out, int out_size, void* d_ws, size_t ws_size,
                              hipStream_t stream) {
    const float* student  = (const float*)d_in[0];   // (8, 512, 2048) f32
    const int*   teacher  = (const int*)  d_in[1];   // (8, 2048)
    const float* codebook = (const float*)d_in[2];   // (4096, 512) f32
    const int*   lengths  = (const int*)  d_in[3];   // (8,)
    float* ws  = (float*)d_ws;
    float* out = (float*)d_out;

    cnorm_kernel<<<NCODES / 8, 256, 0, stream>>>(codebook, ws);

    if (ws_size >= WS_NEED) {
        char* cb_bf = (char*)d_ws + CB_OFF;
        pack_codebook_kernel<<<(NTILES * 512) / 256, 256, 0, stream>>>(
            codebook, (unsigned*)cb_bf);
        triplet_fast<<<(B_BATCH * T_FRAMES) / 128, 256, 0, stream>>>(
            student, teacher, lengths, ws, cb_bf, ws + NCODES);
    } else {
        triplet_fallback<<<(B_BATCH * T_FRAMES) / 128, 256, 0, stream>>>(
            student, teacher, codebook, lengths, ws);
    }

    finalize_kernel<<<1, 64, 0, stream>>>(lengths, ws, out);
}
